// TopKSparseAutoencoder_42442866819908
// MI455X (gfx1250) — matrix operand order repaired and checked by reading, not compiled, not run on hardware
//
#include <hip/hip_runtime.h>
#include <hip/hip_bf16.h>

// Problem constants (from reference): B=8192, D_IN=768, D_HID=16384, K=32
#define BATCH 8192
#define DIN   768
#define DHID  16384
#define TOPK  32

typedef __attribute__((ext_vector_type(16))) __bf16 v16bf;
typedef __attribute__((ext_vector_type(8)))  __bf16 v8bf;
typedef __attribute__((ext_vector_type(8)))  float  v8f;

__device__ __forceinline__ __bf16 f2bf(float f) {
  unsigned u = __builtin_bit_cast(unsigned, f);
  unsigned r = (u + 0x7FFFu + ((u >> 16) & 1u)) >> 16;   // round-to-nearest-even
  unsigned short s = (unsigned short)r;
  return __builtin_bit_cast(__bf16, s);
}

// ---------------------------------------------------------------------------
// Kernel 1: f32 -> bf16 conversion (streaming, 8 elems/thread)
// ---------------------------------------------------------------------------
__global__ __launch_bounds__(256) void cvt_bf16_kernel(
    const float* __restrict__ src, __bf16* __restrict__ dst, int n) {
  int i = (blockIdx.x * 256 + threadIdx.x) * 8;
  if (i >= n) return;
  float4 a = *(const float4*)(src + i);
  float4 b = *(const float4*)(src + i + 4);
  v8bf o;
  o[0] = f2bf(a.x); o[1] = f2bf(a.y); o[2] = f2bf(a.z); o[3] = f2bf(a.w);
  o[4] = f2bf(b.x); o[5] = f2bf(b.y); o[6] = f2bf(b.z); o[7] = f2bf(b.w);
  *(v8bf*)(dst + i) = o;
}

// ---------------------------------------------------------------------------
// Kernel 2: encoder GEMM  pre = x @ W_enc^T + b_enc   (bf16 WMMA, f32 acc)
// One wave -> 16(M) x 128(N) strip = 8 accumulator tiles; K loop of 24 steps.
// ---------------------------------------------------------------------------
__global__ __launch_bounds__(256) void sae_enc_gemm(
    const __bf16* __restrict__ xbf,    // [BATCH, DIN]
    const __bf16* __restrict__ wbf,    // [DHID, DIN]
    const float*  __restrict__ benc,   // [DHID]
    float*        __restrict__ pre) {  // [BATCH, DHID]
  const int wave  = threadIdx.x >> 5;
  const int lane  = threadIdx.x & 31;
  const int gw    = blockIdx.x * 8 + wave;     // 65536 wave-tiles total
  const int mt    = gw >> 7;                   // 0..511  (16-row block)
  const int nb    = gw & 127;                  // 0..127  (128-col block)
  const int m0    = mt << 4;
  const int n0    = nb << 7;
  const int lrow  = lane & 15;
  const int lhalf = lane >> 4;

  const __bf16* arow = xbf + (size_t)(m0 + lrow) * DIN;

  v8f acc[8] = {};

  for (int kb = 0; kb < DIN; kb += 32) {
    // A fragment (16x32 bf16): per-lane K = {kb+8*lhalf..+7} and {kb+16+8*lhalf..+7}
    union { v16bf v; v8bf h[2]; } A;
    A.h[0] = *(const v8bf*)(arow + kb + lhalf * 8);
    A.h[1] = *(const v8bf*)(arow + kb + 16 + lhalf * 8);

#pragma unroll
    for (int t = 0; t < 8; ++t) {
      // B fragment (32x16 bf16): lane = column (W_enc row), K = {kb+8*lhalf..+7} and {kb+16+8*lhalf..+7}
      const __bf16* brow = wbf + (size_t)(n0 + t * 16 + lrow) * DIN + kb + lhalf * 8;
      union { v16bf v; v8bf h[2]; } Bm;
      Bm.h[0] = *(const v8bf*)(brow);
      Bm.h[1] = *(const v8bf*)(brow + 16);
      acc[t] = __builtin_amdgcn_wmma_f32_16x16x32_bf16(
          /*neg_a=*/false, A.v, /*neg_b=*/false, Bm.v,
          /*c_mod=*/(short)0, acc[t], /*reuse_a=*/false, /*reuse_b=*/false);
    }
  }

  // C/D layout: VGPR r, lanes 0-15 -> M=m0+r, lanes 16-31 -> M=m0+8+r; N = n0+t*16+lrow
#pragma unroll
  for (int t = 0; t < 8; ++t) {
    const int n = n0 + t * 16 + lrow;
    const float bias = benc[n];
    float* outp = pre + (size_t)(m0 + lhalf * 8) * DHID + n;
#pragma unroll
    for (int r = 0; r < 8; ++r)
      outp[(size_t)r * DHID] = acc[t][r] + bias;
  }
}

// ---------------------------------------------------------------------------
// Kernel 3: per-row top-32 + sparse rewrite + decode
// One 256-thread block per row. Row values held in registers (64/thread).
// ---------------------------------------------------------------------------
__global__ __launch_bounds__(256) void sae_topk_decode(
    float*       __restrict__ sparse,  // [BATCH, DHID]: pre_act in, sparse_act out
    const float* __restrict__ wdec,    // [DIN, DHID]
    const float* __restrict__ bdec,    // [DIN]
    float*       __restrict__ xrec,    // [BATCH, DIN]
    float*       __restrict__ idxout)  // [BATCH, TOPK] (indices as float)
{
  __shared__ float s_val[8];
  __shared__ int   s_idx[8];
  __shared__ float s_topv[TOPK];
  __shared__ int   s_topi[TOPK];
  __shared__ int   s_bi;

  const int row  = blockIdx.x;
  const int t    = threadIdx.x;
  const int lane = t & 31;
  const int wv   = t >> 5;
  float* prow = sparse + (size_t)row * DHID;

  // Each thread owns 64 contiguous elements: h = t*64 + i
  float v[64];
#pragma unroll
  for (int ii = 0; ii < 16; ++ii) {
    float4 f = *(const float4*)(prow + t * 64 + ii * 4);
    v[ii * 4 + 0] = f.x; v[ii * 4 + 1] = f.y;
    v[ii * 4 + 2] = f.z; v[ii * 4 + 3] = f.w;
  }

  for (int it = 0; it < TOPK; ++it) {
    // local argmax over 64 registers
    float mv = v[0]; int mi = 0;
#pragma unroll
    for (int i = 1; i < 64; ++i)
      if (v[i] > mv) { mv = v[i]; mi = i; }
    int gh = t * 64 + mi;

    // wave32 reduce (max value, tie -> smaller index, matches lax.top_k order)
    for (int off = 16; off > 0; off >>= 1) {
      float ov = __shfl_xor(mv, off, 32);
      int   og = __shfl_xor(gh, off, 32);
      if (ov > mv || (ov == mv && og < gh)) { mv = ov; gh = og; }
    }
    if (lane == 0) { s_val[wv] = mv; s_idx[wv] = gh; }
    __syncthreads();
    if (t == 0) {
      float bv = s_val[0]; int bi = s_idx[0];
      for (int w = 1; w < 8; ++w)
        if (s_val[w] > bv || (s_val[w] == bv && s_idx[w] < bi)) { bv = s_val[w]; bi = s_idx[w]; }
      s_bi = bi; s_topv[it] = bv; s_topi[it] = bi;
    }
    __syncthreads();
    const int bi = s_bi;
    if ((bi >> 6) == t) v[bi & 63] = -__builtin_inff();  // mask out winner
  }
  __syncthreads();

  // Rewrite row: relu(topk) at selected positions, zero elsewhere
#pragma unroll
  for (int ii = 0; ii < 16; ++ii) {
    float4 o;
    float* op = (float*)&o;
#pragma unroll
    for (int q = 0; q < 4; ++q) {
      const int i = ii * 4 + q;
      float outv = 0.0f;
      if (v[i] == -__builtin_inff()) {               // this slot was selected
        const int h = t * 64 + i;
        for (int j = 0; j < TOPK; ++j)
          if (s_topi[j] == h) { outv = fmaxf(s_topv[j], 0.0f); break; }
      }
      op[q] = outv;
    }
    *(float4*)(prow + t * 64 + ii * 4) = o;
  }

  // top-k indices (values stored descending, as in reference)
  if (t < TOPK) idxout[(size_t)row * TOPK + t] = (float)s_topi[t];

  // decode: x_recon[row, i] = b_dec[i] + sum_j relu(v_j) * W_dec[i, idx_j]
#pragma unroll
  for (int c = 0; c < 3; ++c) {
    const int i = t + c * 256;
    float acc = bdec[i];
#pragma unroll
    for (int j = 0; j < TOPK; ++j) {
      const float r = fmaxf(s_topv[j], 0.0f);
      acc += r * wdec[(size_t)i * DHID + s_topi[j]];
    }
    xrec[(size_t)row * DIN + i] = acc;
  }
}

// ---------------------------------------------------------------------------
// Launch: inputs = {x, W_enc, b_enc, W_dec, b_dec, k}
// outputs (flat, in order) = x_recon [B*DIN], sparse_act [B*DHID], topk_idx [B*K]
// ---------------------------------------------------------------------------
extern "C" void kernel_launch(void* const* d_in, const int* in_sizes, int n_in,
                              void* d_out, int out_size, void* d_ws, size_t ws_size,
                              hipStream_t stream) {
  const float* x     = (const float*)d_in[0];
  const float* W_enc = (const float*)d_in[1];
  const float* b_enc = (const float*)d_in[2];
  const float* W_dec = (const float*)d_in[3];
  const float* b_dec = (const float*)d_in[4];
  // d_in[5] = k (always 32 for this problem)

  float* out    = (float*)d_out;
  float* xrec   = out;                                  // [BATCH, DIN]
  float* sparse = out + (size_t)BATCH * DIN;            // [BATCH, DHID]
  float* idxout = sparse + (size_t)BATCH * DHID;        // [BATCH, TOPK]

  __bf16* xbf = (__bf16*)d_ws;                          // 12.6 MB
  __bf16* wbf = xbf + (size_t)BATCH * DIN;              // 25.2 MB

  const int nx = BATCH * DIN;     // 6,291,456  (divisible by 8)
  const int nw = DHID * DIN;      // 12,582,912 (divisible by 8)
  cvt_bf16_kernel<<<(nx / 8 + 255) / 256, 256, 0, stream>>>(x, xbf, nx);
  cvt_bf16_kernel<<<(nw / 8 + 255) / 256, 256, 0, stream>>>(W_enc, wbf, nw);

  // 512 M-blocks x 128 N-blocks = 65536 wave-tiles / 8 waves per block
  sae_enc_gemm<<<8192, 256, 0, stream>>>(xbf, wbf, b_enc, sparse);

  sae_topk_decode<<<BATCH, 256, 0, stream>>>(sparse, W_dec, b_dec, xrec, idxout);
}
